// MySubClassModel_47382079209765
// MI455X (gfx1250) — matrix operand order repaired and checked by reading, not compiled, not run on hardware
//
#include <hip/hip_runtime.h>
#include <hip/hip_bf16.h>
#include <math.h>

// ---------------------------------------------------------------------------
// Model dims
#define BN   128
#define HN   14
#define HW   196            // 14*14
#define CH   512
#define MTOT 25088          // BN*HW
#define FEAT ((size_t)BN * HW * CH)   // 12,845,056
#define TAO  (0.5f / 196.0f)
#define BETA 2.0f

typedef __attribute__((ext_vector_type(16))) __bf16 v16bf;
typedef __attribute__((ext_vector_type(8)))  __bf16 v8bf;
typedef __attribute__((ext_vector_type(8)))  float  v8f;

__device__ __forceinline__ int iabs(int v) { return v < 0 ? -v : v; }

__device__ __forceinline__ float tpl_val(int h, int w, int pi, int pj) {
    int d = iabs(h - pi) + iabs(w - pj);
    float t = 1.0f - BETA * (float)d * (1.0f / 14.0f);
    return TAO * fmaxf(t, -1.0f);
}

// ---------------------------------------------------------------------------
// Fused: per-(b,c) spatial argmax + masked relu + template/raw emission + mean.
// One thread per (b,c) column; lanes cover consecutive c -> coalesced.
// May run in place (src == xout).
__global__ __launch_bounds__(256)
void mask_stats_kernel(const float* __restrict__ src,
                       float* __restrict__ xout,
                       float* __restrict__ tplout,   // may be null
                       float* __restrict__ rawout,   // may be null
                       int*   __restrict__ idxout,   // [BN*CH]
                       float* __restrict__ meansout) // [BN*CH]
{
    int t = blockIdx.x * 256 + threadIdx.x;      // b*512 + c
    int b = t >> 9;
    int c = t & 511;
    const float* col = src + (size_t)b * HW * CH + c;

    // pass 1: argmax (first max wins, matching jnp.argmax)
    float best = -3.402823466e38f;
    int arg = 0;
    for (int hw = 0; hw < HW; ++hw) {
        float v = col[(size_t)hw * CH];
        if (v > best) { best = v; arg = hw; }
    }
    int pi = arg / 14, pj = arg - pi * 14;

    // pass 2: masked = relu(x * tpl), accumulate mean
    float sum = 0.0f;
    int h = 0, w = 0;
    for (int hw = 0; hw < HW; ++hw) {
        float v = col[(size_t)hw * CH];
        float tp = tpl_val(h, w, pi, pj);
        float xm = v * tp;
        xm = xm > 0.0f ? xm : 0.0f;
        size_t off = (size_t)(b * HW + hw) * CH + c;
        xout[off] = xm;
        if (tplout) tplout[off] = tp;
        if (rawout) rawout[off] = v;
        sum += xm;
        if (++w == 14) { w = 0; ++h; }
    }
    idxout[t]   = arg;
    meansout[t] = sum * (1.0f / 196.0f);
}

// ---------------------------------------------------------------------------
// 3x3 SAME conv 512->512 + bias + relu as implicit GEMM with split-precision
// bf16 WMMA: A*B ~= Ahi*Bhi + Ahi*Blo + Alo*Bhi (v_wmma_f32_16x16x32_bf16).
// Block tile 64(M) x 64(N); 8 waves as 4(M) x 2(N); each wave owns 16Mx32N
// (2 accumulators, A-operand reuse). Double-buffered LDS + register prefetch.
#define LDSK 40   // padded k-stride (keeps 16B row alignment: 40*2=80 bytes)

__global__ __launch_bounds__(256)
void conv_wmma_kernel(const float* __restrict__ x1,    // [MTOT][CH]
                      const float* __restrict__ wgt,   // [3][3][CH][CH]
                      const float* __restrict__ bias,  // [CH]
                      float* __restrict__ x3)          // [MTOT][CH]
{
    __shared__ __align__(16) __bf16 Ahi[2][64 * LDSK];
    __shared__ __align__(16) __bf16 Alo[2][64 * LDSK];
    __shared__ __align__(16) __bf16 Bhi[2][64 * LDSK];
    __shared__ __align__(16) __bf16 Blo[2][64 * LDSK];

    const int tid  = threadIdx.x;
    const int lane = tid & 31;
    const int wv   = tid >> 5;
    const int mw   = wv >> 1;           // 0..3  (M sub-tile)
    const int nw   = wv & 1;            // 0..1  (N pair)
    const int nb = blockIdx.x & 7;
    const int mb = blockIdx.x >> 3;
    const int m0 = mb * 64;
    const int n0 = nb * 64;

    // Staging maps: one contiguous 8-float segment per thread (2x b128 loads).
    const int s_ar = tid >> 2;              // A row 0..63
    const int s_aj = (tid & 3) * 8;         // A k-offset {0,8,16,24}
    const int s_bj = tid >> 3;              // B k-row 0..31
    const int s_bi = (tid & 7) * 8;         // B n-offset {0..56}

    // Pre-decode this thread's A row -> (b, oh, ow)
    const int a_m  = m0 + s_ar;
    const int a_b  = a_m / HW;
    const int a_hw = a_m - a_b * HW;
    const int a_oh = a_hw / 14;
    const int a_ow = a_hw - a_oh * 14;

    // WMMA operand selectors (wave32); A and B use the same contiguous k order.
    const int akb  = (lane < 16) ? 0 : 16;  // A: contiguous 16-K halves
    const int bkb  = (lane < 16) ? 0 : 16;  // B: contiguous 16-K halves
    const int arow = mw * 16 + (lane & 15);
    const int brow0 = nw * 32 + (lane & 15);
    const int brow1 = brow0 + 16;

    v8f acc0 = {}, acc1 = {};
    float aF[8], bF[8];

    // ---- chunk prefetch: ck in [0,144): tap = ck/16, ic0 = (ck%16)*32
    auto prefetch = [&](int ck) {
        int tap = ck >> 4;
        int ic0 = (ck & 15) << 5;
        int kh = tap / 3, kw = tap - kh * 3;
        int ih = a_oh + kh - 1, iw = a_ow + kw - 1;
        if ((unsigned)ih < 14u && (unsigned)iw < 14u) {
            const float* p = &x1[((size_t)(a_b * HW + ih * 14 + iw)) * CH + ic0 + s_aj];
            *(float4*)&aF[0] = *(const float4*)p;
            *(float4*)&aF[4] = *(const float4*)(p + 4);
        } else {
            #pragma unroll
            for (int q = 0; q < 8; ++q) aF[q] = 0.0f;
        }
        const float* pw = &wgt[((size_t)(tap * CH + ic0 + s_bj)) * CH + n0 + s_bi];
        *(float4*)&bF[0] = *(const float4*)pw;
        *(float4*)&bF[4] = *(const float4*)(pw + 4);
    };

    prefetch(0);
    int buf = 0;

    for (int ck = 0; ck < 144; ++ck) {
        // ---- commit prefetched regs into LDS[buf] (split hi/lo)
        v8bf ah, al;
        #pragma unroll
        for (int q = 0; q < 8; ++q) {
            __bf16 hv = (__bf16)aF[q];
            ah[q] = hv;
            al[q] = (__bf16)(aF[q] - (float)hv);
        }
        *(v8bf*)&Ahi[buf][s_ar * LDSK + s_aj] = ah;
        *(v8bf*)&Alo[buf][s_ar * LDSK + s_aj] = al;
        #pragma unroll
        for (int q = 0; q < 8; ++q) {   // B transpose scatter into [n][k]
            __bf16 hv = (__bf16)bF[q];
            Bhi[buf][(s_bi + q) * LDSK + s_bj] = hv;
            Blo[buf][(s_bi + q) * LDSK + s_bj] = (__bf16)(bF[q] - (float)hv);
        }
        __syncthreads();

        // ---- overlap: global prefetch of next chunk during compute
        int nk = ck + 1 < 144 ? ck + 1 : 143;
        prefetch(nk);

        // ---- build A operands (shared across both N tiles)
        v8bf x0 = *(const v8bf*)&Ahi[buf][arow * LDSK + akb];
        v8bf x1v = *(const v8bf*)&Ahi[buf][arow * LDSK + akb + 8];
        v8bf y0 = *(const v8bf*)&Alo[buf][arow * LDSK + akb];
        v8bf y1 = *(const v8bf*)&Alo[buf][arow * LDSK + akb + 8];
        v16bf ahi, alo;
        #pragma unroll
        for (int i = 0; i < 8; ++i) {
            ahi[i] = x0[i]; ahi[i + 8] = x1v[i];
            alo[i] = y0[i]; alo[i + 8] = y1[i];
        }
        // ---- N tile 0
        {
            v8bf u0 = *(const v8bf*)&Bhi[buf][brow0 * LDSK + bkb];
            v8bf u1 = *(const v8bf*)&Bhi[buf][brow0 * LDSK + bkb + 8];
            v8bf w0 = *(const v8bf*)&Blo[buf][brow0 * LDSK + bkb];
            v8bf w1 = *(const v8bf*)&Blo[buf][brow0 * LDSK + bkb + 8];
            v16bf bhi, blo;
            #pragma unroll
            for (int i = 0; i < 8; ++i) {
                bhi[i] = u0[i]; bhi[i + 8] = u1[i];
                blo[i] = w0[i]; blo[i + 8] = w1[i];
            }
            acc0 = __builtin_amdgcn_wmma_f32_16x16x32_bf16(
                       false, ahi, false, bhi, (short)0, acc0, false, false);
            acc0 = __builtin_amdgcn_wmma_f32_16x16x32_bf16(
                       false, ahi, false, blo, (short)0, acc0, false, false);
            acc0 = __builtin_amdgcn_wmma_f32_16x16x32_bf16(
                       false, alo, false, bhi, (short)0, acc0, false, false);
        }
        // ---- N tile 1
        {
            v8bf u0 = *(const v8bf*)&Bhi[buf][brow1 * LDSK + bkb];
            v8bf u1 = *(const v8bf*)&Bhi[buf][brow1 * LDSK + bkb + 8];
            v8bf w0 = *(const v8bf*)&Blo[buf][brow1 * LDSK + bkb];
            v8bf w1 = *(const v8bf*)&Blo[buf][brow1 * LDSK + bkb + 8];
            v16bf bhi, blo;
            #pragma unroll
            for (int i = 0; i < 8; ++i) {
                bhi[i] = u0[i]; bhi[i + 8] = u1[i];
                blo[i] = w0[i]; blo[i + 8] = w1[i];
            }
            acc1 = __builtin_amdgcn_wmma_f32_16x16x32_bf16(
                       false, ahi, false, bhi, (short)0, acc1, false, false);
            acc1 = __builtin_amdgcn_wmma_f32_16x16x32_bf16(
                       false, ahi, false, blo, (short)0, acc1, false, false);
            acc1 = __builtin_amdgcn_wmma_f32_16x16x32_bf16(
                       false, alo, false, bhi, (short)0, acc1, false, false);
        }
        buf ^= 1;
    }

    // ---- epilogue: bias + relu, C layout: VGPR r -> M = r + 8*(lane>=16)
    int mbase = m0 + mw * 16 + (lane >> 4) * 8;
    {
        int ncol = n0 + nw * 32 + (lane & 15);
        float bv = bias[ncol];
        #pragma unroll
        for (int r = 0; r < 8; ++r) {
            float v = acc0[r] + bv;
            v = v > 0.0f ? v : 0.0f;
            x3[(size_t)(mbase + r) * CH + ncol] = v;
        }
    }
    {
        int ncol = n0 + nw * 32 + 16 + (lane & 15);
        float bv = bias[ncol];
        #pragma unroll
        for (int r = 0; r < 8; ++r) {
            float v = acc1[r] + bv;
            v = v > 0.0f ? v : 0.0f;
            x3[(size_t)(mbase + r) * CH + ncol] = v;
        }
    }
}

// ---------------------------------------------------------------------------
// act/cls segment sums -> per-channel argmax class. 1 block x 512 threads.
__global__ void stats_kernel(const float* __restrict__ means,    // [BN][CH]
                             const float* __restrict__ act_sums, // [CH][10]
                             const float* __restrict__ cls_sums, // [10]
                             const int*   __restrict__ gt,       // [BN]
                             int* __restrict__ fclass)           // [CH]
{
    __shared__ float clsn[10];
    __shared__ int gts[BN];
    int c = threadIdx.x;
    if (c < BN) gts[c] = gt[c];
    if (c == 0) {
        float cnt[10];
        #pragma unroll
        for (int g = 0; g < 10; ++g) cnt[g] = cls_sums[g];
        for (int b = 0; b < BN; ++b) cnt[gt[b]] += 1.0f;
        #pragma unroll
        for (int g = 0; g < 10; ++g) clsn[g] = cnt[g];
    }
    __syncthreads();

    float acc[10];
    #pragma unroll
    for (int g = 0; g < 10; ++g) acc[g] = act_sums[c * 10 + g];
    for (int b = 0; b < BN; ++b) {
        float m = means[b * CH + c];
        int g0 = gts[b];
        #pragma unroll
        for (int g = 0; g < 10; ++g) acc[g] += (g == g0) ? m : 0.0f;
    }
    float bestv = -3.402823466e38f;
    int bestg = 0;
    #pragma unroll
    for (int g = 0; g < 10; ++g) {
        float fm = (clsn[g] == 0.0f) ? 0.0f : acc[g] / clsn[g];
        if (fm > bestv) { bestv = fm; bestg = g; }
    }
    fclass[c] = bestg;
}

// ---------------------------------------------------------------------------
// loss = match ? relu(template) : 0, templates recomputed from argmax indices.
__global__ __launch_bounds__(256)
void loss_kernel(const int* __restrict__ idx,      // [BN*CH]
                 const int* __restrict__ fclass,   // [CH]
                 const int* __restrict__ gt,       // [BN]
                 float* __restrict__ loss)         // [BN][HW][CH]
{
    int t = blockIdx.x * 256 + threadIdx.x;
    int b = t >> 9, c = t & 511;
    bool match = (fclass[c] == gt[b]);
    int a = idx[t];
    int pi = a / 14, pj = a - pi * 14;
    float* out = loss + (size_t)b * HW * CH + c;
    int h = 0, w = 0;
    for (int hw = 0; hw < HW; ++hw) {
        float v = 0.0f;
        if (match) {
            float tp = tpl_val(h, w, pi, pj);
            v = tp > 0.0f ? tp : 0.0f;
        }
        out[(size_t)hw * CH] = v;
        if (++w == 14) { w = 0; ++h; }
    }
}

// ---------------------------------------------------------------------------
// 2x2 maxpool + dense [25088 -> 10] + softmax. One block per batch row.
__global__ __launch_bounds__(256)
void dense_softmax_kernel(const float* __restrict__ x2,   // [BN][14][14][CH]
                          const float* __restrict__ dw,   // [25088][10]
                          const float* __restrict__ db,   // [10]
                          float* __restrict__ probs)      // [BN][10]
{
    int b = blockIdx.x;
    int t = threadIdx.x;
    const float* xb = x2 + (size_t)b * HW * CH;

    float acc[10] = {};
    for (int n = t; n < 7 * 7 * CH; n += 256) {
        int c  = n & 511;
        int pp = n >> 9;          // ph*7 + pw
        int pw = pp % 7;
        int ph = pp / 7;
        size_t r0 = (size_t)(2 * ph * 14 + 2 * pw) * CH + c;
        float m0 = xb[r0];
        float m1 = xb[r0 + CH];
        float m2 = xb[r0 + 14 * CH];
        float m3 = xb[r0 + 14 * CH + CH];
        float p = fmaxf(fmaxf(m0, m1), fmaxf(m2, m3));
        #pragma unroll
        for (int g = 0; g < 10; ++g) acc[g] += p * dw[(size_t)n * 10 + g];
    }

    __shared__ float red[256];
    float logits[10];
    #pragma unroll
    for (int g = 0; g < 10; ++g) {
        red[t] = acc[g];
        __syncthreads();
        for (int s = 128; s > 0; s >>= 1) {
            if (t < s) red[t] += red[t + s];
            __syncthreads();
        }
        logits[g] = red[0];
        __syncthreads();
    }
    if (t == 0) {
        float mx = -3.402823466e38f;
        #pragma unroll
        for (int g = 0; g < 10; ++g) { logits[g] += db[g]; mx = fmaxf(mx, logits[g]); }
        float s = 0.0f, e[10];
        #pragma unroll
        for (int g = 0; g < 10; ++g) { e[g] = expf(logits[g] - mx); s += e[g]; }
        #pragma unroll
        for (int g = 0; g < 10; ++g) probs[b * 10 + g] = e[g] / s;
    }
}

// ---------------------------------------------------------------------------
extern "C" void kernel_launch(void* const* d_in, const int* in_sizes, int n_in,
                              void* d_out, int out_size, void* d_ws, size_t ws_size,
                              hipStream_t stream) {
    const float* inputs    = (const float*)d_in[0];
    const int*   gt        = (const int*)  d_in[1];
    const float* conv_w    = (const float*)d_in[2];
    const float* conv_b    = (const float*)d_in[3];
    const float* dense_w   = (const float*)d_in[4];
    const float* dense_b   = (const float*)d_in[5];
    const float* act_sums1 = (const float*)d_in[6];
    const float* act_sums2 = (const float*)d_in[7];
    const float* cls_sums1 = (const float*)d_in[8];
    const float* cls_sums2 = (const float*)d_in[9];

    float* out = (float*)d_out;
    float* o_probs = out;                       // [128*10]
    float* o_x1    = out + 1280;                // [FEAT]
    float* o_x2    = o_x1 + FEAT;               // [FEAT] (stages x3, then x2 in place)
    float* o_loss1 = o_x2 + FEAT;
    float* o_loss2 = o_loss1 + FEAT;
    float* o_raw   = o_loss2 + FEAT;
    float* o_post1 = o_raw + FEAT;

    // workspace layout (~1 MB)
    int*   idx1    = (int*)d_ws;
    int*   idx2    = idx1 + BN * CH;
    float* means1  = (float*)(idx2 + BN * CH);
    float* means2  = means1 + BN * CH;
    int*   fclass1 = (int*)(means2 + BN * CH);
    int*   fclass2 = fclass1 + CH;

    const int colBlocks = (BN * CH) / 256;      // 256

    // Stage 1: masked_output on inputs (+ raw copy, pos_t1, means, argmax)
    mask_stats_kernel<<<colBlocks, 256, 0, stream>>>(
        inputs, o_x1, o_post1, o_raw, idx1, means1);
    stats_kernel<<<1, 512, 0, stream>>>(means1, act_sums1, cls_sums1, gt, fclass1);
    loss_kernel<<<colBlocks, 256, 0, stream>>>(idx1, fclass1, gt, o_loss1);

    // Stage 2: conv (WMMA implicit GEMM) -> x3 staged into x2 slot
    conv_wmma_kernel<<<(MTOT / 64) * (CH / 64), 256, 0, stream>>>(
        o_x1, conv_w, conv_b, o_x2);

    // Stage 3: masked_output on x3 (in place -> x2)
    mask_stats_kernel<<<colBlocks, 256, 0, stream>>>(
        o_x2, o_x2, nullptr, nullptr, idx2, means2);
    stats_kernel<<<1, 512, 0, stream>>>(means2, act_sums2, cls_sums2, gt, fclass2);
    loss_kernel<<<colBlocks, 256, 0, stream>>>(idx2, fclass2, gt, o_loss2);

    // Stage 4: maxpool + dense + softmax
    dense_softmax_kernel<<<BN, 256, 0, stream>>>(o_x2, dense_w, dense_b, o_probs);
}
